// OntologyEmbedding_28604482191517
// MI455X (gfx1250) — matrix operand order repaired and checked by reading, not compiled, not run on hardware
//
#include <hip/hip_runtime.h>

typedef __attribute__((ext_vector_type(16))) _Float16 v16h;
typedef __attribute__((ext_vector_type(8)))  _Float16 v8h;
typedef __attribute__((ext_vector_type(8)))  float    v8f;

#define HEADS     5
#define CPH       20
#define NCH       100      // in_channels == heads*out_channels == 100
#define KPAD      128      // K padded for WMMA (4 chunks of 32)
#define NPADC     112      // padded output columns (7 tiles of 16)
#define NTILES    7        // ceil(100/16)
#define NEG_SLOPE 0.2f

// ---------------------------------------------------------------------------
// Utility: float atomic max via monotonic int encodings (handles negatives).
// ---------------------------------------------------------------------------
__device__ __forceinline__ void atomicMaxFloat(float* addr, float val) {
    if (val >= 0.0f) {
        atomicMax((int*)addr, __float_as_int(val));
    } else {
        atomicMin((unsigned int*)addr, __float_as_uint(val));
    }
}

__global__ void zero_kernel(float* __restrict__ p, int n) {
    int i = blockIdx.x * blockDim.x + threadIdx.x;
    if (i < n) p[i] = 0.0f;
}

// ---------------------------------------------------------------------------
// Prep: convert+pad X (nrows x 100 f32) -> Xh (mpad x 128 f16, zero padded)
// ---------------------------------------------------------------------------
__global__ void cvt_X(const float* __restrict__ X, _Float16* __restrict__ Xh,
                      int nrows, int mpad) {
    int i = blockIdx.x * blockDim.x + threadIdx.x;
    if (i >= mpad * KPAD) return;
    const int row = i >> 7;          // /128
    const int k   = i & (KPAD - 1);
    float v = (row < nrows && k < NCH) ? X[(size_t)row * NCH + k] : 0.0f;
    Xh[i] = (_Float16)v;
}

// Prep: convert+pad+transpose W (100x100 f32, row-major [k][col])
//       -> WhT (112 x 128 f16, [col][k])
__global__ void cvt_W(const float* __restrict__ W, _Float16* __restrict__ WhT) {
    int i = blockIdx.x * blockDim.x + threadIdx.x;
    if (i >= NPADC * KPAD) return;
    const int col = i >> 7;
    const int k   = i & (KPAD - 1);
    float v = (col < NCH && k < NCH) ? W[(size_t)k * NCH + col] : 0.0f;
    WhT[i] = (_Float16)v;
}

// ---------------------------------------------------------------------------
// Dense projection H = X @ W via v_wmma_f32_16x16x32_f16, one 16x16 tile/wave.
// Fragment layouts per CDNA5 ISA 7.12.2 (wave32):
//   A 16x32 f16 : lane L holds row M=L&15; elements 0..7  = K kb+hiHalf*8 .. +7
//                                          elements 8..15 = K kb+16+hiHalf*8 .. +7
//   B 32x16 f16 : lane L holds col N=L&15; elements 0..7  = K kb+hiHalf*8 .. +7
//                                          elements 8..15 = K kb+16+hiHalf*8 .. +7
//   C/D 16x16 f32: vgpr r -> row = r + hiHalf*8, col = L&15
// All loads are unconditional contiguous vector loads from the padded f16 copies.
// ---------------------------------------------------------------------------
__global__ void gemm_wmma(const _Float16* __restrict__ Xh, const _Float16* __restrict__ WhT,
                          float* __restrict__ Hout, int nrows, int mtiles) {
    const int lane  = threadIdx.x & 31;
    const int gwave = blockIdx.x * (blockDim.x >> 5) + (threadIdx.x >> 5);
    if (gwave >= mtiles * NTILES) return;        // wave-uniform exit
    const int mtile  = gwave / NTILES;
    const int ntile  = gwave % NTILES;
    const int r15    = lane & 15;
    const int hiHalf = lane >> 4;
    const int ncol   = ntile * 16 + r15;         // < 112 always

    const _Float16* xrow = Xh  + (size_t)(mtile * 16 + r15) * KPAD;
    const _Float16* wcol = WhT + (size_t)ncol * KPAD;

    v8f acc = {};
    #pragma unroll
    for (int kb = 0; kb < KPAD; kb += 32) {
        const v8h  a0 = *(const v8h*)(xrow + kb + hiHalf * 8);
        const v8h  a1 = *(const v8h*)(xrow + kb + 16 + hiHalf * 8);
        const v8h  b0 = *(const v8h*)(wcol + kb + hiHalf * 8);
        const v8h  b1 = *(const v8h*)(wcol + kb + 16 + hiHalf * 8);
        v16h a, b;
        #pragma unroll
        for (int t = 0; t < 8; ++t) { a[t] = a0[t]; a[8 + t] = a1[t]; b[t] = b0[t]; b[8 + t] = b1[t]; }
        acc = __builtin_amdgcn_wmma_f32_16x16x32_f16(
            /*neg_a=*/false, a, /*neg_b=*/false, b,
            /*c_mod=*/(short)0, acc, /*reuse_a=*/false, /*reuse_b=*/false);
    }

    // Store: row-range check is wave-uniform (whole tile in range or tail tile).
    const int  row0     = mtile * 16 + hiHalf * 8;
    const bool fullTile = (mtile * 16 + 16 <= nrows);
    float*     outp     = Hout + (size_t)row0 * NCH + ncol;
    if (fullTile) {
        if (ncol < NCH) {                         // single divergent lane mask
            #pragma unroll
            for (int r = 0; r < 8; ++r) outp[(size_t)r * NCH] = acc[r];
        }
    } else {
        #pragma unroll
        for (int r = 0; r < 8; ++r) {
            if (ncol < NCH && row0 + r < nrows) outp[(size_t)r * NCH] = acc[r];
        }
    }
}

// ---------------------------------------------------------------------------
// Per-node attention scalars + softmax state init.
//   a_dst[n,h] = <h[n,h,:], att[h,0:20]>,  a_src[n,h] = <h[n,h,:], att[h,20:40]>
// ---------------------------------------------------------------------------
__global__ void node_attn_init(const float* __restrict__ Hbuf, const float* __restrict__ att,
                               float* __restrict__ a_dst, float* __restrict__ a_src,
                               float* __restrict__ segmax, float* __restrict__ denom, int n) {
    int idx = blockIdx.x * blockDim.x + threadIdx.x;
    if (idx >= n * HEADS) return;
    const int node = idx / HEADS;
    const int hd   = idx % HEADS;
    const float* hp = Hbuf + (size_t)node * NCH + hd * CPH;
    const float* ap = att + hd * (2 * CPH);
    float s1 = 0.0f, s2 = 0.0f;
    #pragma unroll
    for (int c = 0; c < CPH; ++c) {
        s1 += hp[c] * ap[c];
        s2 += hp[c] * ap[CPH + c];
    }
    a_dst[idx]  = s1;
    a_src[idx]  = s2;
    segmax[idx] = -__builtin_inff();
    denom[idx]  = 0.0f;
}

__device__ __forceinline__ void edge_endpoints(int e, int E, const int* __restrict__ ed,
                                               const int* __restrict__ es, int* d, int* s) {
    if (e < E) { *d = ed[e]; *s = es[e]; }
    else       { *d = *s = e - E; }          // appended self-loops
}

// Pass 1: segment max of leaky-relu logits over destination.
__global__ void edge_max(const int* __restrict__ ed, const int* __restrict__ es, int E, int n,
                         const float* __restrict__ a_dst, const float* __restrict__ a_src,
                         float* __restrict__ segmax) {
    int e = blockIdx.x * blockDim.x + threadIdx.x;
    if (e >= E + n) return;
    int d, s;
    edge_endpoints(e, E, ed, es, &d, &s);
    #pragma unroll
    for (int hd = 0; hd < HEADS; ++hd) {
        float l = a_dst[d * HEADS + hd] + a_src[s * HEADS + hd];
        l = (l >= 0.0f) ? l : NEG_SLOPE * l;
        atomicMaxFloat(&segmax[d * HEADS + hd], l);
    }
}

// Pass 2: denom = segment sum of exp(logit - max).
__global__ void edge_sum(const int* __restrict__ ed, const int* __restrict__ es, int E, int n,
                         const float* __restrict__ a_dst, const float* __restrict__ a_src,
                         const float* __restrict__ segmax, float* __restrict__ denom) {
    int e = blockIdx.x * blockDim.x + threadIdx.x;
    if (e >= E + n) return;
    int d, s;
    edge_endpoints(e, E, ed, es, &d, &s);
    #pragma unroll
    for (int hd = 0; hd < HEADS; ++hd) {
        float l = a_dst[d * HEADS + hd] + a_src[s * HEADS + hd];
        l = (l >= 0.0f) ? l : NEG_SLOPE * l;
        atomicAdd(&denom[d * HEADS + hd], __expf(l - segmax[d * HEADS + hd]));
    }
}

// Pass 3: out[dst,h,:] += alpha * h[src,h,:]   (one thread per edge*head, 20 atomics)
__global__ void edge_agg(const int* __restrict__ ed, const int* __restrict__ es, int E, int n,
                         const float* __restrict__ a_dst, const float* __restrict__ a_src,
                         const float* __restrict__ segmax, const float* __restrict__ denom,
                         const float* __restrict__ Hbuf, float* __restrict__ out) {
    int idx = blockIdx.x * blockDim.x + threadIdx.x;
    if (idx >= (E + n) * HEADS) return;
    const int e  = idx / HEADS;
    const int hd = idx % HEADS;
    int d, s;
    edge_endpoints(e, E, ed, es, &d, &s);
    float l = a_dst[d * HEADS + hd] + a_src[s * HEADS + hd];
    l = (l >= 0.0f) ? l : NEG_SLOPE * l;
    const float alpha = __expf(l - segmax[d * HEADS + hd]) / denom[d * HEADS + hd];
    const float* hs = Hbuf + (size_t)s * NCH + hd * CPH;
    float*       od = out  + (size_t)d * NCH + hd * CPH;
    #pragma unroll
    for (int c = 0; c < CPH; ++c) {
        atomicAdd(&od[c], hs[c] * alpha);
    }
}

__global__ void bias_add(float* __restrict__ Y, const float* __restrict__ bias, int total) {
    int i = blockIdx.x * blockDim.x + threadIdx.x;
    if (i < total) Y[i] += bias[i % NCH];
}

__global__ void gather_out(const float* __restrict__ Y, const int* __restrict__ idxmap,
                           float* __restrict__ out, int total) {
    int i = blockIdx.x * blockDim.x + threadIdx.x;
    if (i >= total) return;
    const int m = i / NCH;
    const int j = i % NCH;
    out[i] = Y[(size_t)idxmap[m] * NCH + j];
}

// ---------------------------------------------------------------------------
// Host-side orchestration
// ---------------------------------------------------------------------------
static inline int ceil_div(int a, int b) { return (a + b - 1) / b; }

static void run_layer(const float* X, const int* edges, int E, int n, int mtiles,
                      const float* att, const float* bias,
                      _Float16* Xh, const _Float16* WhT,
                      float* Hbuf, float* Yout,
                      float* a_dst, float* a_src, float* segmax, float* denom,
                      hipStream_t stream) {
    const int* ed   = edges;         // edge_index[0] = dst
    const int* es   = edges + E;     // edge_index[1] = src
    const int  mpad = mtiles * 16;

    // 1) convert/pad input to f16, then H = X @ W (WMMA)
    cvt_X<<<ceil_div(mpad * KPAD, 256), 256, 0, stream>>>(X, Xh, n, mpad);
    {
        const int waves  = mtiles * NTILES;
        const int blocks = ceil_div(waves, 8);          // 256 threads = 8 waves
        gemm_wmma<<<blocks, 256, 0, stream>>>(Xh, WhT, Hbuf, n, mtiles);
    }
    // 2) per-node attention scalars + softmax state init
    node_attn_init<<<ceil_div(n * HEADS, 256), 256, 0, stream>>>(
        Hbuf, att, a_dst, a_src, segmax, denom, n);
    // 3) zero aggregation target (may alias X; GEMM above already consumed X)
    zero_kernel<<<ceil_div(n * NCH, 256), 256, 0, stream>>>(Yout, n * NCH);
    // 4) softmax passes + aggregation
    const int EP = E + n;
    edge_max<<<ceil_div(EP, 256), 256, 0, stream>>>(ed, es, E, n, a_dst, a_src, segmax);
    edge_sum<<<ceil_div(EP, 256), 256, 0, stream>>>(ed, es, E, n, a_dst, a_src, segmax, denom);
    edge_agg<<<ceil_div(EP * HEADS, 256), 256, 0, stream>>>(
        ed, es, E, n, a_dst, a_src, segmax, denom, Hbuf, Yout);
    // 5) bias
    bias_add<<<ceil_div(n * NCH, 256), 256, 0, stream>>>(Yout, bias, n * NCH);
}

extern "C" void kernel_launch(void* const* d_in, const int* in_sizes, int n_in,
                              void* d_out, int out_size, void* d_ws, size_t ws_size,
                              hipStream_t stream) {
    const float* embedding = (const float*)d_in[0];
    const float* weight    = (const float*)d_in[1];
    const float* att       = (const float*)d_in[2];
    const float* bias      = (const float*)d_in[3];
    const int*   edges1    = (const int*)d_in[4];
    const int*   edges2    = (const int*)d_in[5];
    const int*   idxmap    = (const int*)d_in[6];

    const int N      = in_sizes[0] / NCH;       // 50000
    const int E      = in_sizes[4] / 2;         // 800000
    const int M      = in_sizes[6];             // 40000
    const int mtiles = ceil_div(N, 16);
    const int mpad   = mtiles * 16;

    // Workspace layout:
    //   f32: Hbuf[N*100], Ybuf[N*100], a_dst/a_src/segmax/denom[N*HEADS each]
    //   f16: Xh[mpad*128], WhT[112*128]
    float* ws     = (float*)d_ws;
    float* Hbuf   = ws;
    float* Ybuf   = Hbuf + (size_t)N * NCH;
    float* a_dst  = Ybuf + (size_t)N * NCH;
    float* a_src  = a_dst + (size_t)N * HEADS;
    float* segmax = a_src + (size_t)N * HEADS;
    float* denom  = segmax + (size_t)N * HEADS;
    _Float16* Xh  = (_Float16*)(denom + (size_t)N * HEADS);
    _Float16* WhT = Xh + (size_t)mpad * KPAD;

    // Weight is shared by both layers: convert once.
    cvt_W<<<ceil_div(NPADC * KPAD, 256), 256, 0, stream>>>(weight, WhT);

    // Layer 1: embedding -> Ybuf
    run_layer(embedding, edges1, E, N, mtiles, att, bias, Xh, WhT,
              Hbuf, Ybuf, a_dst, a_src, segmax, denom, stream);
    // Layer 2: Ybuf -> Ybuf (cvt_X snapshots Ybuf into Xh before it is re-zeroed)
    run_layer(Ybuf, edges2, E, N, mtiles, att, bias, Xh, WhT,
              Hbuf, Ybuf, a_dst, a_src, segmax, denom, stream);

    // Final gather by idx_mapping
    gather_out<<<ceil_div(M * NCH, 256), 256, 0, stream>>>(
        Ybuf, idxmap, (float*)d_out, M * NCH);
}
